// OriginGNNv5_6468220748390
// MI455X (gfx1250) — matrix operand order repaired and checked by reading, not compiled, not run on hardware
//
#include <hip/hip_runtime.h>
#include <stdint.h>

// ---------------------------------------------------------------------------
// Types
// ---------------------------------------------------------------------------
typedef __attribute__((ext_vector_type(16))) __bf16   v16bf;
typedef __attribute__((ext_vector_type(8)))  float    v8f;
typedef __attribute__((ext_vector_type(4)))  uint32_t u32x4;
typedef unsigned short ushort_t;

#define NH 128
#define NA_CNT 50000
#define NO_CNT 10000
#define NG_CNT 50000
#define E_ANA_CNT 800000
#define E_ONA_CNT 200000
#define E_TA_CNT  50000

// float -> bf16 (round-to-nearest-even)
static __device__ __forceinline__ ushort_t f2bf(float f) {
    uint32_t u = __float_as_uint(f);
    u += 0x7FFFu + ((u >> 16) & 1u);
    return (ushort_t)(u >> 16);
}

// Load a 16-element bf16 fragment as two 16-byte chunks (ds_load_b128 x2)
static __device__ __forceinline__ v16bf ld_frag2(const ushort_t* p0, const ushort_t* p1) {
    union { v16bf v; u32x4 q[2]; } u;
    u.q[0] = *(const u32x4*)p0;
    u.q[1] = *(const u32x4*)p1;
    return u.v;
}

// float atomic-max via signed/unsigned integer ordering trick.
// Requires destination pre-initialized to -inf (0xFF800000).
static __device__ __forceinline__ void atomicMaxF(float* addr, float v) {
    if (v >= 0.0f) atomicMax((int*)addr, __float_as_int(v));
    else           atomicMin((unsigned int*)addr, __float_as_uint(v));
}

// Per-lane async 16-byte copy global -> LDS (ASYNCcnt tracked)
static __device__ __forceinline__ void async_g2l_b128(uint32_t lds_addr, const void* gaddr) {
    asm volatile("global_load_async_to_lds_b128 %0, %1, off"
                 :: "v"(lds_addr), "v"(gaddr) : "memory");
}
static __device__ __forceinline__ void wait_asynccnt0() {
    asm volatile("s_wait_asynccnt 0x0" ::: "memory");
}

// ---------------------------------------------------------------------------
// Node embedding: out[i,:] = x[i,:din] @ W[din,128] + b  (f32 + bf16 mirror)
// ---------------------------------------------------------------------------
__global__ void node_embed_kernel(const float* __restrict__ x, int n, int din,
                                  const float* __restrict__ W, const float* __restrict__ b,
                                  float* __restrict__ out, ushort_t* __restrict__ out_bf) {
    int total = n * NH;
    for (int idx = blockIdx.x * blockDim.x + threadIdx.x; idx < total;
         idx += gridDim.x * blockDim.x) {
        int i = idx >> 7, c = idx & 127;
        float v = b[c];
        const float* xr = x + (size_t)i * din;
        for (int k = 0; k < din; ++k) v += xr[k] * W[k * NH + c];
        out[idx] = v;
        out_bf[idx] = f2bf(v);
    }
}

// ---------------------------------------------------------------------------
// Edge embedding: e = relu(ea @ W1[4,128] + b1) @ W2[128,128] + b2, stored bf16.
// ---------------------------------------------------------------------------
#define EE_LDS_BYTES (128*136*2 /*w2s*/ + 128*136*2 /*hid*/ + 512*4 /*w1e*/ + 128*4 + 128*4 + 512*4 /*eas*/)

__global__ void __launch_bounds__(256)
edge_embed_kernel(const float* __restrict__ ea, int E,
                  const float* __restrict__ W1, const float* __restrict__ b1,
                  const float* __restrict__ W2, const float* __restrict__ b2,
                  ushort_t* __restrict__ eout) {
    extern __shared__ __align__(16) char smem[];
    ushort_t* w2s = (ushort_t*)smem;              // 128*136 halves
    ushort_t* hid = w2s + 128 * 136;              // 128*136 halves
    float*    w1e = (float*)(hid + 128 * 136);    // 512
    float*    b1s = w1e + 512;                    // 128
    float*    b2s = b1s + 128;                    // 128
    float*    eas = b2s + 128;                    // 512

    const int tid = threadIdx.x;
    const int wave = tid >> 5, lane = tid & 31;
    const int g = lane >> 4, n16 = lane & 15;

    for (int i = tid; i < 128 * 128; i += 256) {
        int k = i >> 7, n = i & 127;
        w2s[n * 136 + k] = f2bf(W2[i]);
    }
    for (int i = tid; i < 512; i += 256) w1e[i] = W1[i];
    if (tid < 128) { b1s[tid] = b1[tid]; b2s[tid] = b2[tid]; }
    __syncthreads();

    const int ntiles = (E + 127) >> 7;
    for (int t = blockIdx.x; t < ntiles; t += gridDim.x) {
        const int base = t << 7;
        for (int i = tid; i < 512; i += 256) {
            size_t gi = (size_t)base * 4 + i;
            eas[i] = (gi < (size_t)E * 4) ? ea[gi] : 0.0f;
        }
        __syncthreads();
        for (int i = tid; i < 128 * 128; i += 256) {
            int r = i >> 7, c = i & 127;
            float v = b1s[c];
#pragma unroll
            for (int k = 0; k < 4; ++k) v += eas[r * 4 + k] * w1e[k * NH + c];
            hid[r * 136 + c] = f2bf(v > 0.0f ? v : 0.0f);
        }
        __syncthreads();

        // GEMM2: each wave owns 16 rows
        const int mb = wave;
        v8f acc[8] = {};
#pragma unroll
        for (int ks = 0; ks < 4; ++ks) {
            const int Koff = ks * 32;
            const ushort_t* ap = hid + (mb * 16 + n16) * 136 + Koff + g * 8;
            v16bf a = ld_frag2(ap, ap + 16);
            v16bf bfr[8];
#pragma unroll
            for (int nb = 0; nb < 8; ++nb) {
                const ushort_t* bp = w2s + (nb * 16 + n16) * 136 + Koff + g * 8;
                bfr[nb] = ld_frag2(bp, bp + 16);
            }
#pragma unroll
            for (int nb = 0; nb < 8; ++nb)
                acc[nb] = __builtin_amdgcn_wmma_f32_16x16x32_bf16(
                    false, a, false, bfr[nb], (short)0, acc[nb], false, false);
        }
#pragma unroll
        for (int nb = 0; nb < 8; ++nb) {
            const int col = nb * 16 + n16;
            const float bb = b2s[col];
#pragma unroll
            for (int r = 0; r < 8; ++r) {
                const int row = mb * 16 + g * 8 + r;
                const int e = base + row;
                if (e < E) eout[(size_t)e * NH + col] = f2bf(acc[nb][r] + bb);
            }
        }
        __syncthreads();
    }
}

// ---------------------------------------------------------------------------
// Conv layer (per edge type), 64-edge tiles, double-buffered async gather:
//   msg tile = [xa_bf[dst] | xsrc_bf[src] | eemb]   (3 x 256B per edge row)
//   h = relu(m @ W1[384,128] + b1); o = h @ W2[128,128] + b2
//   agg[dst] = atomic-max(agg[dst], o)
// LDS = 100352(W1) + 34816(W2) + 2*50176(msg) + 17408(hid) + 1024(b) + 512(idx)
//     = 254464 B  (< 320KB/WGP)
// Wave w: M-block = w>>1 (16 edges), column half = w&1 (64 cols)
// ---------------------------------------------------------------------------
#define CONV_LDS_BYTES (128*392*2 + 128*136*2 + 2*64*392*2 + 64*136*2 + 128*4 + 128*4 + 2*64*4)

__global__ void __launch_bounds__(256)
conv_kernel(const ushort_t* __restrict__ xdst_bf, const ushort_t* __restrict__ xsrc_bf,
            const ushort_t* __restrict__ eemb, const int* __restrict__ ei, int E,
            const float* __restrict__ W1g, const float* __restrict__ b1g,
            const float* __restrict__ W2g, const float* __restrict__ b2g,
            float* __restrict__ agg) {
    extern __shared__ __align__(16) char smem[];
    ushort_t* w1s  = (ushort_t*)smem;               // [n:128][K:384] stride 392
    ushort_t* w2s  = w1s + 128 * 392;               // [n:128][K:128] stride 136
    ushort_t* msgb = w2s + 128 * 136;               // 2 x [row:64][K:384] stride 392
    ushort_t* hid  = msgb + 2 * 64 * 392;           // [row:64][K:128] stride 136
    float*    b1s  = (float*)(hid + 64 * 136);
    float*    b2s  = b1s + 128;
    int*      dstl = (int*)(b2s + 128);             // [2][64]

    const int tid = threadIdx.x;
    const int wave = tid >> 5, lane = tid & 31;
    const int g = lane >> 4, n16 = lane & 15;

    // stage weights (swizzled panel-major, padded stride: conflict-free frags)
    for (int i = tid; i < 384 * 128; i += 256) {
        int k = i >> 7, n = i & 127;
        w1s[n * 392 + k] = f2bf(W1g[i]);
    }
    for (int i = tid; i < 128 * 128; i += 256) {
        int k = i >> 7, n = i & 127;
        w2s[n * 136 + k] = f2bf(W2g[i]);
    }
    if (tid < 128) { b1s[tid] = b1g[tid]; b2s[tid] = b2g[tid]; }
    __syncthreads();

    const int* __restrict__ srcI = ei;
    const int* __restrict__ dstI = ei + E;
    const int ntiles = (E + 63) >> 6;

    // issue async gather of tile t into buffer bi
    auto issue_tile = [&](int t, int bi) {
        const int base = t << 6;
        ushort_t* mb_ = msgb + bi * 64 * 392;
        const uint32_t mlds = (uint32_t)(uintptr_t)(void*)mb_;
        if (tid < 64) {
            int e = base + tid;
            dstl[bi * 64 + tid] = (e < E) ? dstI[e] : 0;
        }
        const int r0 = wave * 8;
#pragma unroll
        for (int rr = 0; rr < 8; ++rr) {
            const int r = r0 + rr;
            const int e = base + r;
            if (e < E) {
                const int d = dstI[e], s = srcI[e];
                // lanes 0-15 -> dst segment, lanes 16-31 -> src segment
                const char* gp0 = (g == 0)
                    ? (const char*)(xdst_bf + (size_t)d * NH)
                    : (const char*)(xsrc_bf + (size_t)s * NH);
                async_g2l_b128(mlds + (uint32_t)(r * 784 + g * 256 + n16 * 16),
                               gp0 + n16 * 16);
                if (g == 0) {
                    const char* gp1 = (const char*)(eemb + (size_t)e * NH) + n16 * 16;
                    async_g2l_b128(mlds + (uint32_t)(r * 784 + 512 + n16 * 16), gp1);
                }
            } else {
                uint32_t* zr = (uint32_t*)(mb_ + r * 392);
                for (int c = lane; c < 196; c += 32) zr[c] = 0u;
            }
        }
    };

    const int mb = wave >> 1;        // M-block (16 edges)
    const int nbs = (wave & 1) * 4;  // first of 4 N-blocks (64 cols)

    int t = blockIdx.x;
    if (t < ntiles) issue_tile(t, 0);
    int cur = 0;

    for (; t < ntiles; t += gridDim.x) {
        const int base = t << 6;
        wait_asynccnt0();        // my async fills for buf[cur] done
        __syncthreads();         // everyone's fills done -> buf[cur] ready

        const int tn = t + gridDim.x;
        if (tn < ntiles) {
            issue_tile(tn, cur ^ 1);   // overlap next gather with compute
            const int t2 = tn + gridDim.x;
            if (t2 < ntiles) {         // warm index cache two tiles ahead
                __builtin_prefetch(dstI + (t2 << 6), 0, 0);
                __builtin_prefetch(srcI + (t2 << 6), 0, 0);
            }
        }

        ushort_t* mbuf = msgb + cur * 64 * 392;
        const int* dstc = dstl + cur * 64;

        // ----- GEMM1: [16 x 384] @ [384 x 64] per wave -----
        v8f acc[4] = {};
#pragma unroll
        for (int ks = 0; ks < 12; ++ks) {
            const int Koff = ks * 32;
            const ushort_t* ap = mbuf + (mb * 16 + n16) * 392 + Koff + g * 8;
            v16bf a = ld_frag2(ap, ap + 16);
            v16bf bfr[4];
#pragma unroll
            for (int j = 0; j < 4; ++j) {
                const ushort_t* bp = w1s + ((nbs + j) * 16 + n16) * 392 + Koff + g * 8;
                bfr[j] = ld_frag2(bp, bp + 16);
            }
#pragma unroll
            for (int j = 0; j < 4; ++j)
                acc[j] = __builtin_amdgcn_wmma_f32_16x16x32_bf16(
                    false, a, false, bfr[j], (short)0, acc[j], false, false);
        }
        // bias + relu -> bf16 hidden (two waves fill disjoint column halves)
#pragma unroll
        for (int j = 0; j < 4; ++j) {
            const int col = (nbs + j) * 16 + n16;
            const float bb = b1s[col];
#pragma unroll
            for (int r = 0; r < 8; ++r) {
                float v = acc[j][r] + bb;
                hid[(mb * 16 + g * 8 + r) * 136 + col] = f2bf(v > 0.0f ? v : 0.0f);
            }
        }
        __syncthreads();   // hidden rows complete across column halves

        // ----- GEMM2: [16 x 128] @ [128 x 64] per wave -----
        v8f acc2[4] = {};
#pragma unroll
        for (int ks = 0; ks < 4; ++ks) {
            const int Koff = ks * 32;
            const ushort_t* ap = hid + (mb * 16 + n16) * 136 + Koff + g * 8;
            v16bf a = ld_frag2(ap, ap + 16);
            v16bf bfr[4];
#pragma unroll
            for (int j = 0; j < 4; ++j) {
                const ushort_t* bp = w2s + ((nbs + j) * 16 + n16) * 136 + Koff + g * 8;
                bfr[j] = ld_frag2(bp, bp + 16);
            }
#pragma unroll
            for (int j = 0; j < 4; ++j)
                acc2[j] = __builtin_amdgcn_wmma_f32_16x16x32_bf16(
                    false, a, false, bfr[j], (short)0, acc2[j], false, false);
        }
        // scatter-max
#pragma unroll
        for (int j = 0; j < 4; ++j) {
            const int col = (nbs + j) * 16 + n16;
            const float bb = b2s[col];
#pragma unroll
            for (int r = 0; r < 8; ++r) {
                const int row = mb * 16 + g * 8 + r;
                const int e = base + row;
                if (e < E) {
                    atomicMaxF(agg + (size_t)dstc[row] * NH + col, acc2[j][r] + bb);
                }
            }
        }
        cur ^= 1;
    }
}

// ---------------------------------------------------------------------------
// Init agg buffers to -inf
// ---------------------------------------------------------------------------
__global__ void init_agg_kernel(unsigned int* __restrict__ p, size_t n) {
    for (size_t i = blockIdx.x * (size_t)blockDim.x + threadIdx.x; i < n;
         i += (size_t)gridDim.x * blockDim.x)
        p[i] = 0xFF800000u;  // -inf
}

// ---------------------------------------------------------------------------
// xa += max( fix(agg0), fix(agg1), fix(agg2) ), fix(-inf) = 0 ; update mirror
// ---------------------------------------------------------------------------
__global__ void combine_kernel(float* __restrict__ xa, ushort_t* __restrict__ xa_bf,
                               const float* __restrict__ a0,
                               const float* __restrict__ a1,
                               const float* __restrict__ a2, int n) {
    for (int i = blockIdx.x * blockDim.x + threadIdx.x; i < n;
         i += gridDim.x * blockDim.x) {
        float v0 = a0[i], v1 = a1[i], v2 = a2[i];
        v0 = (__float_as_uint(v0) == 0xFF800000u) ? 0.0f : v0;
        v1 = (__float_as_uint(v1) == 0xFF800000u) ? 0.0f : v1;
        v2 = (__float_as_uint(v2) == 0xFF800000u) ? 0.0f : v2;
        float nv = xa[i] + fmaxf(v0, fmaxf(v1, v2));
        xa[i] = nv;
        xa_bf[i] = f2bf(nv);
    }
}

// ---------------------------------------------------------------------------
// Field head: out[i] = relu(cat(xa[i],act[i]) @ W1[131,128] + b1) @ W2[128] + b2
// ---------------------------------------------------------------------------
__global__ void __launch_bounds__(128)
field_kernel(const float* __restrict__ xa, const float* __restrict__ act,
             const float* __restrict__ W1, const float* __restrict__ b1,
             const float* __restrict__ W2, const float* __restrict__ b2,
             float* __restrict__ out, int n) {
    __shared__ float hrow[132];
    __shared__ float red[128];
    const int t = threadIdx.x;
    for (int i = blockIdx.x; i < n; i += gridDim.x) {
        hrow[t] = xa[(size_t)i * NH + t];
        if (t < 3) hrow[128 + t] = act[(size_t)i * 3 + t];
        __syncthreads();
        float v = b1[t];
        for (int k = 0; k < 131; ++k) v += hrow[k] * W1[k * NH + t];
        v = v > 0.0f ? v : 0.0f;
        red[t] = v * W2[t];
        __syncthreads();
        for (int s = 64; s > 0; s >>= 1) {
            if (t < s) red[t] += red[t + s];
            __syncthreads();
        }
        if (t == 0) out[i] = red[0] + b2[0];
        __syncthreads();
    }
}

// ---------------------------------------------------------------------------
// Host-side launch
// ---------------------------------------------------------------------------
extern "C" void kernel_launch(void* const* d_in, const int* in_sizes, int n_in,
                              void* d_out, int out_size, void* d_ws, size_t ws_size,
                              hipStream_t stream) {
    (void)in_sizes; (void)n_in; (void)out_size; (void)ws_size;

    const float* x_agent    = (const float*)d_in[0];
    const float* x_obstacle = (const float*)d_in[1];
    const float* x_goal     = (const float*)d_in[2];
    const float* ea_ana     = (const float*)d_in[3];
    const float* ea_ona     = (const float*)d_in[4];
    const float* ea_ta      = (const float*)d_in[5];
    const float* action     = (const float*)d_in[6];
    const int*   ei_ana     = (const int*)d_in[7];
    const int*   ei_ona     = (const int*)d_in[8];
    const int*   ei_ta      = (const int*)d_in[9];
    const float* Wa = (const float*)d_in[10]; const float* ba = (const float*)d_in[11];
    const float* Wo = (const float*)d_in[12]; const float* bo = (const float*)d_in[13];
    const float* Wg = (const float*)d_in[14]; const float* bg = (const float*)d_in[15];
    // edge_embed: 16..27 ; convs: 28..63 ; field: 64..67
    const float* fW1 = (const float*)d_in[64]; const float* fb1 = (const float*)d_in[65];
    const float* fW2 = (const float*)d_in[66]; const float* fb2 = (const float*)d_in[67];

    size_t off = 0;
    auto carve = [&](size_t bytes) -> char* {
        char* p = (char*)d_ws + off;
        off += (bytes + 255) & ~(size_t)255;
        return p;
    };
    float* xa = (float*)carve((size_t)NA_CNT * NH * 4);
    float* xo = (float*)carve((size_t)NO_CNT * NH * 4);
    float* xg = (float*)carve((size_t)NG_CNT * NH * 4);
    ushort_t* xa_bf = (ushort_t*)carve((size_t)NA_CNT * NH * 2);
    ushort_t* xo_bf = (ushort_t*)carve((size_t)NO_CNT * NH * 2);
    ushort_t* xg_bf = (ushort_t*)carve((size_t)NG_CNT * NH * 2);
    float* agg0 = (float*)carve((size_t)NA_CNT * NH * 4);
    float* agg1 = (float*)carve((size_t)NA_CNT * NH * 4);
    float* agg2 = (float*)carve((size_t)NA_CNT * NH * 4);
    ushort_t* e_ana = (ushort_t*)carve((size_t)E_ANA_CNT * NH * 2);
    ushort_t* e_ona = (ushort_t*)carve((size_t)E_ONA_CNT * NH * 2);
    ushort_t* e_ta  = (ushort_t*)carve((size_t)E_TA_CNT  * NH * 2);

    // ---- node embeddings (f32 + bf16 mirror)
    {
        int blk = 256;
        node_embed_kernel<<<(NA_CNT * NH + blk - 1) / blk, blk, 0, stream>>>(
            x_agent, NA_CNT, 6, Wa, ba, xa, xa_bf);
        node_embed_kernel<<<(NO_CNT * NH + blk - 1) / blk, blk, 0, stream>>>(
            x_obstacle, NO_CNT, 4, Wo, bo, xo, xo_bf);
        node_embed_kernel<<<(NG_CNT * NH + blk - 1) / blk, blk, 0, stream>>>(
            x_goal, NG_CNT, 6, Wg, bg, xg, xg_bf);
    }

    // ---- edge embeddings (bf16, reused by all layers)
    {
        int tiles;
        tiles = (E_ANA_CNT + 127) / 128;
        edge_embed_kernel<<<tiles < 4096 ? tiles : 4096, 256, EE_LDS_BYTES, stream>>>(
            ea_ana, E_ANA_CNT,
            (const float*)d_in[16], (const float*)d_in[17],
            (const float*)d_in[18], (const float*)d_in[19], e_ana);
        tiles = (E_ONA_CNT + 127) / 128;
        edge_embed_kernel<<<tiles < 4096 ? tiles : 4096, 256, EE_LDS_BYTES, stream>>>(
            ea_ona, E_ONA_CNT,
            (const float*)d_in[20], (const float*)d_in[21],
            (const float*)d_in[22], (const float*)d_in[23], e_ona);
        tiles = (E_TA_CNT + 127) / 128;
        edge_embed_kernel<<<tiles < 4096 ? tiles : 4096, 256, EE_LDS_BYTES, stream>>>(
            ea_ta, E_TA_CNT,
            (const float*)d_in[24], (const float*)d_in[25],
            (const float*)d_in[26], (const float*)d_in[27], e_ta);
    }

    // ---- 3 conv layers
    const size_t agg_elems = (size_t)3 * NA_CNT * NH;
    for (int l = 0; l < 3; ++l) {
        init_agg_kernel<<<4096, 256, 0, stream>>>((unsigned int*)agg0, agg_elems);

        const int pb = 28 + l * 12;
        int tiles;

        tiles = (E_ANA_CNT + 63) / 64;
        conv_kernel<<<tiles < 4096 ? tiles : 4096, 256, CONV_LDS_BYTES, stream>>>(
            xa_bf, xa_bf, e_ana, ei_ana, E_ANA_CNT,
            (const float*)d_in[pb + 0], (const float*)d_in[pb + 1],
            (const float*)d_in[pb + 2], (const float*)d_in[pb + 3], agg0);

        tiles = (E_ONA_CNT + 63) / 64;
        conv_kernel<<<tiles < 4096 ? tiles : 4096, 256, CONV_LDS_BYTES, stream>>>(
            xa_bf, xo_bf, e_ona, ei_ona, E_ONA_CNT,
            (const float*)d_in[pb + 4], (const float*)d_in[pb + 5],
            (const float*)d_in[pb + 6], (const float*)d_in[pb + 7], agg1);

        tiles = (E_TA_CNT + 63) / 64;
        conv_kernel<<<tiles < 4096 ? tiles : 4096, 256, CONV_LDS_BYTES, stream>>>(
            xa_bf, xg_bf, e_ta, ei_ta, E_TA_CNT,
            (const float*)d_in[pb + 8], (const float*)d_in[pb + 9],
            (const float*)d_in[pb + 10], (const float*)d_in[pb + 11], agg2);

        combine_kernel<<<(NA_CNT * NH + 255) / 256, 256, 0, stream>>>(
            xa, xa_bf, agg0, agg1, agg2, NA_CNT * NH);
    }

    // ---- field head
    field_kernel<<<NA_CNT, 128, 0, stream>>>(
        xa, action, fW1, fb1, fW2, fb2, (float*)d_out, NA_CNT);
}
